// SarvamMoESparseMoeBlock_73847667687620
// MI455X (gfx1250) — compile-verified, matrix operand order checked by reading, not run on hardware
//
#include <hip/hip_runtime.h>
#include <cstdint>
#include <cstddef>

#define T_TOK 1024
#define DIM   2048
#define NEXP  32
#define IEXP  1024
#define SHI   2048
#define TOPK  8
#define NGRP  8
#define TKG   4
#define GSZ   (NEXP / NGRP)
#define RSCALE 2.5f

typedef __attribute__((ext_vector_type(16))) __bf16 v16bf;
typedef __attribute__((ext_vector_type(2)))  __bf16 v2bf;
typedef __attribute__((ext_vector_type(8)))  float  v8f;

union V16U { v16bf v; unsigned int u[8]; };

// fp32 -> packed bf16x2; native cast lets clang pick v_cvt_pk_bf16_f32 if present
__device__ __forceinline__ unsigned int packbf2(float a, float b) {
  union { v2bf v; unsigned int u; } r;
  r.v[0] = (__bf16)a;
  r.v[1] = (__bf16)b;
  return r.u;
}
__device__ __forceinline__ unsigned short f2bf(float a) {
  union { __bf16 h; unsigned short s; } r;
  r.h = (__bf16)a;
  return r.s;
}
__device__ __forceinline__ v8f zero8() {
  v8f z = {0.f, 0.f, 0.f, 0.f, 0.f, 0.f, 0.f, 0.f};
  return z;
}

// ---------------------------------------------------------------- router ----
__global__ void zero_cnt_kernel(int* cnt) {
  if (threadIdx.x < NEXP) cnt[threadIdx.x] = 0;
}

__global__ __launch_bounds__(32)
void router_kernel(const float* __restrict__ x, const float* __restrict__ gw,
                   const float* __restrict__ bias, unsigned short* __restrict__ Xb,
                   int* __restrict__ cnt, int* __restrict__ perm, float* __restrict__ pw,
                   int* __restrict__ tok_e, int* __restrict__ tok_pos) {
  const int t = blockIdx.x;
  __shared__ __align__(16) float xs[DIM];
  __shared__ float sc[NEXP];
  const int lane = threadIdx.x;
  for (int k = lane * 4; k < DIM; k += 32 * 4) {
    float4 v = *(const float4*)(x + (size_t)t * DIM + k);
    *(float4*)&xs[k] = v;
    unsigned int p0 = packbf2(v.x, v.y), p1 = packbf2(v.z, v.w);
    *(uint2*)(Xb + (size_t)t * DIM + k) = make_uint2(p0, p1);
  }
  __syncthreads();
  float acc = 0.f;
  const float4* wv = (const float4*)(gw + (size_t)lane * DIM);
  for (int k = 0; k < DIM / 4; ++k) {
    float4 w4 = wv[k];
    float4 x4 = *(const float4*)&xs[k * 4];
    acc += w4.x * x4.x + w4.y * x4.y + w4.z * x4.z + w4.w * x4.w;
  }
  sc[lane] = 1.f / (1.f + __expf(-acc));
  __syncthreads();
  if (lane == 0) {
    float sch[NEXP];
    for (int e2 = 0; e2 < NEXP; ++e2) sch[e2] = sc[e2] + bias[e2];
    float gs[NGRP];
    for (int g = 0; g < NGRP; ++g) {
      float m1 = -1e30f, m2 = -1e30f;
      for (int j = 0; j < GSZ; ++j) {
        float v = sch[g * GSZ + j];
        if (v > m1) { m2 = m1; m1 = v; }
        else if (v > m2) m2 = v;
      }
      gs[g] = m1 + m2;
    }
    bool gsel[NGRP];
    for (int g = 0; g < NGRP; ++g) gsel[g] = false;
    for (int it = 0; it < TKG; ++it) {
      float best = -1e30f; int bg = 0;
      for (int g = 0; g < NGRP; ++g)
        if (!gsel[g] && gs[g] > best) { best = gs[g]; bg = g; }
      gsel[bg] = true;
    }
    bool taken[NEXP];
    for (int e2 = 0; e2 < NEXP; ++e2) taken[e2] = false;
    int ids[TOPK]; float wts[TOPK]; float wsum = 0.f;
    for (int it = 0; it < TOPK; ++it) {
      float best = -1e30f; int be = 0;
      for (int e2 = 0; e2 < NEXP; ++e2)
        if (gsel[e2 / GSZ] && !taken[e2] && sch[e2] > best) { best = sch[e2]; be = e2; }
      taken[be] = true; ids[it] = be; wts[it] = sc[be]; wsum += sc[be];
    }
    float inv = RSCALE / wsum;
    for (int it = 0; it < TOPK; ++it) {
      int e2 = ids[it];
      int pos = atomicAdd(&cnt[e2], 1);
      perm[e2 * T_TOK + pos] = t;
      pw[e2 * T_TOK + pos]   = wts[it] * inv;
      tok_e[t * TOPK + it]   = e2;
      tok_pos[t * TOPK + it] = pos;
    }
  }
}

__global__ void scan_kernel(const int* __restrict__ cnt, int* __restrict__ basep) {
  if (threadIdx.x == 0 && blockIdx.x == 0) {
    int s = 0;
    for (int e = 0; e < NEXP; ++e) { basep[e] = s; s += cnt[e]; }
  }
}

// ----------------------------------------------------- gate_up + SiLU*up ----
// Tile 64 rows x 128 i-features; 8 waves 2(M)x4(N); double-buffered LDS.
template <bool GATHER>
__global__ __launch_bounds__(256)
void gateup_wmma_kernel(const unsigned short* __restrict__ Xb,   // bf16 [T, D]
                        const float* __restrict__ Wg,            // [2*Ieff, D] (+e)
                        const int* __restrict__ perm, const float* __restrict__ pw,
                        const int* __restrict__ cnt, const int* __restrict__ basep,
                        unsigned short* __restrict__ Hout,       // bf16 [rows, Ieff]
                        int Ieff) {
  const int e = GATHER ? (int)blockIdx.z : 0;
  const int m0 = (int)blockIdx.y * 64;
  const int mcount = GATHER ? cnt[e] : T_TOK;
  if (m0 >= mcount) return;
  const int mloc = (mcount - m0 < 64) ? (mcount - m0) : 64;
  const int n0 = (int)blockIdx.x * 128;
  const float* W = Wg + (GATHER ? (size_t)e * (size_t)(2 * IEXP) * DIM : (size_t)0);
  const int hbase = (GATHER ? basep[e] : 0) + m0;

  __shared__ __align__(16) unsigned int Ash[2][64][16];   // double-buffered
  __shared__ __align__(16) unsigned int Bsh[2][256][16];
  __shared__ int   rowIdx[64];
  __shared__ float rowW[64];

  const int tid = threadIdx.x;
  if (tid < 64) {
    // invalid rows clamp to a valid token: their C rows are masked in epilogue
    int idx = m0 + ((tid < mloc) ? tid : 0);
    rowIdx[tid] = GATHER ? perm[e * T_TOK + idx] : idx;
    rowW[tid]   = (GATHER && tid < mloc) ? pw[e * T_TOK + m0 + tid] : 1.0f;
  }
  __syncthreads();

  const int wid  = tid >> 5;
  const int lane = tid & 31;
  const int half = lane >> 4;
  const int r16  = lane & 15;
  const int mh   = wid >> 2;       // wave M half
  const int nq   = wid & 3;        // wave N quarter
  const int arow = tid >> 2;       // A staging row (0..63)
  const int au4  = (tid & 3) * 4;  // A staging uint4 col
  const int bu   = (tid & 7) * 2;  // B staging packed-uint col pair

  // hoisted source pointers (branchless staging)
  const unsigned int* aSrc =
      (const unsigned int*)(Xb + (size_t)rowIdx[arow] * DIM) + au4;
  const float* bp[8];
#pragma unroll
  for (int rep = 0; rep < 8; ++rep) {
    int brow = (tid >> 3) + rep * 32;
    int n = n0 + (brow & 127);
    size_t wrow = (brow < 128) ? (size_t)n : (size_t)(Ieff + n);
    bp[rep] = W + wrow * (size_t)DIM + bu * 2;
  }

  v8f accg[2][2], accu[2][2];
#pragma unroll
  for (int mi = 0; mi < 2; ++mi)
#pragma unroll
    for (int ni = 0; ni < 2; ++ni) { accg[mi][ni] = zero8(); accu[mi][ni] = zero8(); }

  auto stage = [&](int b, int k0) {
    *(uint4*)&Ash[b][arow][au4] = *(const uint4*)(aSrc + (k0 >> 1));
#pragma unroll
    for (int rep = 0; rep < 8; ++rep) {
      int brow = (tid >> 3) + rep * 32;
      float4 f = *(const float4*)(bp[rep] + k0);
      Bsh[b][brow][bu]     = packbf2(f.x, f.y);
      Bsh[b][brow][bu + 1] = packbf2(f.z, f.w);
    }
  };

  stage(0, 0);
  int buf = 0;
  for (int k0 = 0; k0 < DIM; k0 += 32) {
    __syncthreads();
    if (k0 + 32 < DIM) stage(buf ^ 1, k0 + 32);

    V16U a[2], bg[2], bup[2];
#pragma unroll
    for (int j = 0; j < 8; ++j) {
      int ku = ((j < 4) ? j : (j + 4)) + 4 * half;   // 16-bit WMMA operand layout
      int ar0 = mh * 32 + r16;
      a[0].u[j] = Ash[buf][ar0][ku];
      a[1].u[j] = Ash[buf][ar0 + 16][ku];
      int br0 = nq * 32 + r16;
      bg[0].u[j]  = Bsh[buf][br0][ku];
      bg[1].u[j]  = Bsh[buf][br0 + 16][ku];
      bup[0].u[j] = Bsh[buf][128 + br0][ku];
      bup[1].u[j] = Bsh[buf][128 + br0 + 16][ku];
    }
#pragma unroll
    for (int mi = 0; mi < 2; ++mi)
#pragma unroll
      for (int ni = 0; ni < 2; ++ni) {
        accg[mi][ni] = __builtin_amdgcn_wmma_f32_16x16x32_bf16(
            false, a[mi].v, false, bg[ni].v, (short)0, accg[mi][ni], false, false);
        accu[mi][ni] = __builtin_amdgcn_wmma_f32_16x16x32_bf16(
            false, a[mi].v, false, bup[ni].v, (short)0, accu[mi][ni], false, false);
      }
    buf ^= 1;
  }

  // epilogue: h = silu(g)*u * weight -> bf16 H (fast rcp, fine for bf16 out)
#pragma unroll
  for (int mi = 0; mi < 2; ++mi)
#pragma unroll
    for (int ni = 0; ni < 2; ++ni) {
      int i = n0 + nq * 32 + ni * 16 + r16;
#pragma unroll
      for (int v = 0; v < 8; ++v) {
        int m = mh * 32 + mi * 16 + v + 8 * half;
        if (m < mloc) {
          float gv = accg[mi][ni][v];
          float uv = accu[mi][ni][v];
          float sg = __builtin_amdgcn_rcpf(1.f + __expf(-gv));
          float h = gv * sg * uv * rowW[m];
          Hout[(size_t)(hbase + m) * Ieff + i] = f2bf(h);
        }
      }
    }
}

// ----------------------------------------------------------- down-project ---
// Tile 64 rows x 128 cols of DIM; 8 waves 2(M)x4(N); double-buffered LDS.
template <bool GATHER>
__global__ __launch_bounds__(256)
void down_wmma_kernel(const unsigned short* __restrict__ Hsrc,  // bf16 [rows, K]
                      const float* __restrict__ Wd,             // [DIM, K] (+e)
                      const int* __restrict__ cnt, const int* __restrict__ basep,
                      float* __restrict__ outp,                 // [rows, DIM]
                      int K) {
  const int e = GATHER ? (int)blockIdx.z : 0;
  const int m0 = (int)blockIdx.y * 64;
  const int mcount = GATHER ? cnt[e] : T_TOK;
  if (m0 >= mcount) return;
  const int mloc = (mcount - m0 < 64) ? (mcount - m0) : 64;
  const int n0 = (int)blockIdx.x * 128;
  const float* W = Wd + (GATHER ? (size_t)e * (size_t)DIM * IEXP : (size_t)0);
  const int hrow0 = (GATHER ? basep[e] : 0) + m0;

  __shared__ __align__(16) unsigned int Ash[2][64][16];
  __shared__ __align__(16) unsigned int Bsh[2][128][16];

  const int tid  = threadIdx.x;
  const int wid  = tid >> 5;
  const int lane = tid & 31;
  const int half = lane >> 4;
  const int r16  = lane & 15;
  const int mh   = wid >> 2;
  const int nq   = wid & 3;
  const int arow = tid >> 2;
  const int au4  = (tid & 3) * 4;
  const int bu   = (tid & 7) * 2;

  // clamp invalid A rows to row 0 of this tile (outputs masked in epilogue)
  const int arClamped = (arow < mloc) ? arow : 0;
  const unsigned int* aSrc =
      (const unsigned int*)(Hsrc + (size_t)(hrow0 + arClamped) * K) + au4;
  const float* bp[4];
#pragma unroll
  for (int rep = 0; rep < 4; ++rep) {
    int n = n0 + (tid >> 3) + rep * 32;
    bp[rep] = W + (size_t)n * K + bu * 2;
  }

  v8f acc[2][2];
#pragma unroll
  for (int mi = 0; mi < 2; ++mi)
#pragma unroll
    for (int ni = 0; ni < 2; ++ni) acc[mi][ni] = zero8();

  auto stage = [&](int b, int k0) {
    *(uint4*)&Ash[b][arow][au4] = *(const uint4*)(aSrc + (k0 >> 1));
#pragma unroll
    for (int rep = 0; rep < 4; ++rep) {
      int brow = (tid >> 3) + rep * 32;
      float4 f = *(const float4*)(bp[rep] + k0);
      Bsh[b][brow][bu]     = packbf2(f.x, f.y);
      Bsh[b][brow][bu + 1] = packbf2(f.z, f.w);
    }
  };

  stage(0, 0);
  int buf = 0;
  for (int k0 = 0; k0 < K; k0 += 32) {
    __syncthreads();
    if (k0 + 32 < K) stage(buf ^ 1, k0 + 32);

    V16U a[2], b[2];
#pragma unroll
    for (int j = 0; j < 8; ++j) {
      int ku = ((j < 4) ? j : (j + 4)) + 4 * half;
      int ar0 = mh * 32 + r16;
      a[0].u[j] = Ash[buf][ar0][ku];
      a[1].u[j] = Ash[buf][ar0 + 16][ku];
      int br0 = nq * 32 + r16;
      b[0].u[j] = Bsh[buf][br0][ku];
      b[1].u[j] = Bsh[buf][br0 + 16][ku];
    }
#pragma unroll
    for (int mi = 0; mi < 2; ++mi)
#pragma unroll
      for (int ni = 0; ni < 2; ++ni)
        acc[mi][ni] = __builtin_amdgcn_wmma_f32_16x16x32_bf16(
            false, a[mi].v, false, b[ni].v, (short)0, acc[mi][ni], false, false);
    buf ^= 1;
  }

#pragma unroll
  for (int mi = 0; mi < 2; ++mi)
#pragma unroll
    for (int ni = 0; ni < 2; ++ni) {
      int n = n0 + nq * 32 + ni * 16 + r16;
#pragma unroll
      for (int v = 0; v < 8; ++v) {
        int m = mh * 32 + mi * 16 + v + 8 * half;
        if (m < mloc)
          outp[(size_t)(hrow0 + m) * DIM + n] = acc[mi][ni][v];
      }
    }
}

// --------------------------------------------------------------- combine ----
__global__ __launch_bounds__(256)
void combine_kernel(const float* __restrict__ Yr, const int* __restrict__ tok_e,
                    const int* __restrict__ tok_pos, const int* __restrict__ basep,
                    float* __restrict__ out) {
  const int t = blockIdx.x;
  __shared__ int slots[TOPK];
  if (threadIdx.x < TOPK) {
    int e = tok_e[t * TOPK + threadIdx.x];
    slots[threadIdx.x] = basep[e] + tok_pos[t * TOPK + threadIdx.x];
  }
  __syncthreads();
  for (int n = threadIdx.x; n < DIM; n += 256) {
    float acc = out[(size_t)t * DIM + n];
#pragma unroll
    for (int k = 0; k < TOPK; ++k) acc += Yr[(size_t)slots[k] * DIM + n];
    out[(size_t)t * DIM + n] = acc;
  }
}

// ---------------------------------------------------------------- launch ----
extern "C" void kernel_launch(void* const* d_in, const int* in_sizes, int n_in,
                              void* d_out, int out_size, void* d_ws, size_t ws_size,
                              hipStream_t stream) {
  (void)in_sizes; (void)n_in; (void)out_size; (void)ws_size;
  const float* x    = (const float*)d_in[0];
  const float* gw   = (const float*)d_in[1];
  const float* bias = (const float*)d_in[2];
  const float* wgu  = (const float*)d_in[3];
  const float* wdn  = (const float*)d_in[4];
  const float* wgus = (const float*)d_in[5];
  const float* wdns = (const float*)d_in[6];
  float* out = (float*)d_out;

  char* ws = (char*)d_ws;
  size_t off = 0;
  auto alloc = [&](size_t bytes) {
    void* p = ws + off;
    off += (bytes + 255) & ~(size_t)255;
    return p;
  };
  unsigned short* Xb  = (unsigned short*)alloc((size_t)T_TOK * DIM * 2);
  unsigned short* Hb  = (unsigned short*)alloc((size_t)T_TOK * TOPK * IEXP * 2);
  unsigned short* Hsb = (unsigned short*)alloc((size_t)T_TOK * SHI * 2);
  float* Yr    = (float*)alloc((size_t)T_TOK * TOPK * DIM * 4);
  int*   cnt   = (int*)alloc(NEXP * 4);
  int*   basep = (int*)alloc(NEXP * 4);
  int*   perm  = (int*)alloc((size_t)NEXP * T_TOK * 4);
  float* pw    = (float*)alloc((size_t)NEXP * T_TOK * 4);
  int*   tokE  = (int*)alloc((size_t)T_TOK * TOPK * 4);
  int*   tokP  = (int*)alloc((size_t)T_TOK * TOPK * 4);

  zero_cnt_kernel<<<1, 32, 0, stream>>>(cnt);
  router_kernel<<<T_TOK, 32, 0, stream>>>(x, gw, bias, Xb, cnt, perm, pw, tokE, tokP);
  scan_kernel<<<1, 32, 0, stream>>>(cnt, basep);
  // routed experts: sparse-dispatch grouped GEMMs
  gateup_wmma_kernel<true><<<dim3(IEXP / 128, T_TOK / 64, NEXP), 256, 0, stream>>>(
      Xb, wgu, perm, pw, cnt, basep, Hb, IEXP);
  down_wmma_kernel<true><<<dim3(DIM / 128, T_TOK / 64, NEXP), 256, 0, stream>>>(
      Hb, wdn, cnt, basep, Yr, IEXP);
  // shared expert MLP (writes out directly)
  gateup_wmma_kernel<false><<<dim3(SHI / 128, T_TOK / 64, 1), 256, 0, stream>>>(
      Xb, wgus, nullptr, nullptr, nullptr, nullptr, Hsb, SHI);
  down_wmma_kernel<false><<<dim3(DIM / 128, T_TOK / 64, 1), 256, 0, stream>>>(
      Hsb, wdns, nullptr, nullptr, out, SHI);
  // deterministic fixed-order sum of the 8 routed contributions + shared
  combine_kernel<<<T_TOK, 256, 0, stream>>>(Yr, tokE, tokP, basep, out);
}
